// BlockMOE_24653112279332
// MI455X (gfx1250) — matrix operand order repaired and checked by reading, not compiled, not run on hardware
//
#include <hip/hip_runtime.h>

// ---------------------------------------------------------------------------
// Problem constants (match reference)
// ---------------------------------------------------------------------------
constexpr int Bb  = 4;
constexpr int Tt  = 1024;
constexpr int Cc  = 1024;
constexpr int NHh = 16;
constexpr int HDd = 64;
constexpr int Ee  = 8;
constexpr int Kk  = 2;
constexpr int Hh  = 2730;
constexpr int CAPp = 640;          // ceil(1.25*B*T/E)
constexpr int Nrows = Bb * Tt;     // 4096
constexpr int NKf = Nrows * Kk;    // 8192

typedef __attribute__((ext_vector_type(16))) __bf16 v16bf;
typedef __attribute__((ext_vector_type(8)))  float  v8f;

// gfx1250 async global->LDS copy (ASYNCcnt path); fallback if builtin absent.
#if defined(__AMDGCN__) && __has_builtin(__builtin_amdgcn_global_load_async_to_lds_b128)
#define HAVE_ASYNC_LDS 1
// Exact parameter types per hipcc diagnostic: AS1/AS3 pointers to gcc-style
// vector_size(16) int vectors.
typedef int v4i_t __attribute__((vector_size(16)));
typedef __attribute__((address_space(1))) v4i_t* gptr128_t;
typedef __attribute__((address_space(3))) v4i_t* lptr128_t;
#else
#define HAVE_ASYNC_LDS 0
#endif

__device__ inline v8f vz8() {
  v8f v;
#pragma unroll
  for (int i = 0; i < 8; ++i) v[i] = 0.f;
  return v;
}

// ---------------------------------------------------------------------------
// Generic bf16-WMMA GEMM:  C = epilogue(A @ op(B)) [+bias] [SiLU] [+Res]
//   A: [M,K] f32 row-major (lda) -- staged to LDS as f32 (async path capable)
//   B: BKMAJOR=false -> [N,K] row-major (ldb=K stride)   (x @ W.T weights)
//      BKMAJOR=true  -> [K,N] row-major (ldb=N stride)   (MoE w1/w2)
//   ASYNCA: use global_load_async_to_lds_b128 for the A tile. Requires
//           M%128==0, K%32==0, 16B-aligned A rows (lda%4==0). Otherwise use
//           the guarded manual path.
// Block: 256 threads (8 waves), tile 128x128, BK=32.
// Each wave: 32(M) x 64(N) -> 2x4 wmma accumulators.
// ---------------------------------------------------------------------------
constexpr int BM = 128, BN = 128, BK = 32;
constexpr int LDSF   = 36;    // f32 A-tile LDS stride (144B rows, 16B aligned)
constexpr int LDS_STR = 40;   // bf16 B-tile LDS stride (80B rows, 16B aligned)

template<bool BKMAJOR, bool SILU, bool ASYNCA>
__global__ __launch_bounds__(256)
void gemm_wmma_kernel(const float* __restrict__ A, const float* __restrict__ Bm,
                      const float* __restrict__ bias, const float* __restrict__ Res,
                      float* __restrict__ Cout,
                      int M, int N, int Kd, int lda, int ldb, int ldc,
                      long aStride, long bStride, long biasStride, long cStride)
{
  __shared__ float  lA[BM * LDSF];      // 18.0 KB (f32 A tile)
  __shared__ __bf16 lB[BN * LDS_STR];   // 10.0 KB (bf16 B tile, [n][k])

  const int e = blockIdx.z;
  A    += (long)e * aStride;
  Bm   += (long)e * bStride;
  Cout += (long)e * cStride;
  if (bias) bias += (long)e * biasStride;

  const int tid  = threadIdx.x;
  const int lane = tid & 31;
  const int wid  = tid >> 5;
  const int wm   = wid >> 1;            // 0..3
  const int wn   = wid & 1;             // 0..1
  const int m0   = blockIdx.y * BM;
  const int n0   = blockIdx.x * BN;
  const int mlane = lane & 15;
  const int kbA  = (lane < 16) ? 0 : 16;
  const int kbB  = (lane < 16) ? 0 : 16;

  v8f acc[2][4];
#pragma unroll
  for (int i = 0; i < 2; ++i)
#pragma unroll
    for (int j = 0; j < 4; ++j) acc[i][j] = vz8();

  for (int k0 = 0; k0 < Kd; k0 += BK) {
    // ---- stage A tile (f32) ----
#if HAVE_ASYNC_LDS
    if constexpr (ASYNCA) {
      // 128x32 f32 = 1024 x 16B chunks, 4 per thread, no guards needed
#pragma unroll
      for (int i = 0; i < (BM * BK) / (4 * 256); ++i) {
        int ci = tid + i * 256;
        int r  = ci >> 3;               // 8 chunks per 32-float row
        int cq = ci & 7;
        const float* gp = A + (long)(m0 + r) * lda + k0 + cq * 4;
        float* lp = &lA[r * LDSF + cq * 4];
        __builtin_amdgcn_global_load_async_to_lds_b128(
            (gptr128_t)(void*)gp, (lptr128_t)(void*)lp, 0, 0);
      }
      asm volatile("s_wait_asynccnt 0" ::: "memory");
    } else
#endif
    {
#pragma unroll
      for (int i = 0; i < (BM * BK) / 256; ++i) {
        int idx = tid + i * 256;
        int r = idx >> 5, c = idx & 31;
        int gr = m0 + r, gc = k0 + c;
        lA[r * LDSF + c] = (gr < M && gc < Kd) ? A[(long)gr * lda + gc] : 0.f;
      }
    }
    // ---- stage B tile as bf16 lB[n][k] ----
    if constexpr (BKMAJOR) {
#pragma unroll
      for (int i = 0; i < (BN * BK) / 256; ++i) {
        int idx = tid + i * 256;
        int k = idx >> 7, n = idx & 127;       // coalesced along N
        int gk = k0 + k, gn = n0 + n;
        float v = (gk < Kd && gn < N) ? Bm[(long)gk * ldb + gn] : 0.f;
        lB[n * LDS_STR + k] = (__bf16)v;
      }
    } else {
#pragma unroll
      for (int i = 0; i < (BN * BK) / 256; ++i) {
        int idx = tid + i * 256;
        int n = idx >> 5, k = idx & 31;        // coalesced along K
        int gn = n0 + n, gk = k0 + k;
        float v = (gn < N && gk < Kd) ? Bm[(long)gn * ldb + gk] : 0.f;
        lB[n * LDS_STR + k] = (__bf16)v;
      }
    }
    // ---- prefetch next K-tile into caches (global_prefetch_b8) ----
    if (k0 + BK < Kd) {
      if (tid < 128) {
        __builtin_prefetch(A + (long)(m0 + tid) * lda + (k0 + BK), 0, 3);
      } else {
        int t = tid - 128;
        if constexpr (BKMAJOR) {
          if (t < 32 && k0 + BK + t < Kd)
            __builtin_prefetch(Bm + (long)(k0 + BK + t) * ldb + n0, 0, 3);
        } else {
          if (n0 + t < N)
            __builtin_prefetch(Bm + (long)(n0 + t) * ldb + (k0 + BK), 0, 3);
        }
      }
    }
    __syncthreads();

    // ---- fragments ----
    v16bf af[2], bfrag[4];
#pragma unroll
    for (int tm = 0; tm < 2; ++tm) {
      int row = wm * 32 + tm * 16 + mlane;
#pragma unroll
      for (int j = 0; j < 8; ++j) {
        af[tm][j]     = (__bf16)lA[row * LDSF + kbA + j];
        af[tm][j + 8] = (__bf16)lA[row * LDSF + kbA + 8 + j];
      }
    }
#pragma unroll
    for (int tn = 0; tn < 4; ++tn) {
      int nrow = wn * 64 + tn * 16 + mlane;
#pragma unroll
      for (int j = 0; j < 16; ++j)
        bfrag[tn][j] = lB[nrow * LDS_STR + kbB + j];
    }
#pragma unroll
    for (int tm = 0; tm < 2; ++tm)
#pragma unroll
      for (int tn = 0; tn < 4; ++tn)
        acc[tm][tn] = __builtin_amdgcn_wmma_f32_16x16x32_bf16(
            false, af[tm], false, bfrag[tn], (short)0, acc[tm][tn], false, false);
    __syncthreads();
  }

  // ---- epilogue ----
#pragma unroll
  for (int tm = 0; tm < 2; ++tm) {
#pragma unroll
    for (int tn = 0; tn < 4; ++tn) {
      int col = n0 + wn * 64 + tn * 16 + mlane;
#pragma unroll
      for (int r = 0; r < 8; ++r) {
        int row = m0 + wm * 32 + tm * 16 + r + ((lane < 16) ? 0 : 8);
        if (row < M && col < N) {
          float v = acc[tm][tn][r];
          if (bias) v += bias[col];
          if constexpr (SILU) v = v / (1.f + __expf(-v));
          if (Res) v += Res[(long)row * ldc + col];
          Cout[(long)row * ldc + col] = v;
        }
      }
    }
  }
}

// ---------------------------------------------------------------------------
// Flash attention (causal). qkv: [B*T, 3C] fp32.  y: [B*T, C] fp32.
// Block = 256 thr (8 waves); each wave owns 16 query rows of one (b,h).
// K operand loaded directly (contiguous 64B per lane, lane-pairs share lines);
// V tile staged per-wave in LDS with coalesced float4 loads.
// grid = (T/128, B*NH)
// ---------------------------------------------------------------------------
__global__ __launch_bounds__(256)
void attn_kernel(const float* __restrict__ qkv, float* __restrict__ y)
{
  __shared__ __bf16 pstg[8][16][32];    // 8 KB  P staging (C-layout -> A-layout)
  __shared__ __bf16 vstg[8][32][72];    // 36 KB per-wave V tile [key][col], padded

  const int tid  = threadIdx.x;
  const int lane = tid & 31;
  const int wid  = tid >> 5;
  const int bh = blockIdx.y;
  const int b  = bh >> 4;
  const int h  = bh & 15;
  const int q0 = blockIdx.x * 128 + wid * 16;
  const int mlane = lane & 15;
  const int kbA = (lane < 16) ? 0 : 16;
  const int kbB = (lane < 16) ? 0 : 16;
  const int rbase = (lane < 16) ? 0 : 8;
  const long rowB = (long)b * Tt;
  const int qoff = h * HDd;
  const int koff = Cc + h * HDd;
  const int voff = 2 * Cc + h * HDd;
  const float scale = 0.125f;           // 1/sqrt(64)

  // Q fragments (head dim 64 -> 2 K-steps), pre-scaled
  v16bf qf[2];
  {
    long qrow = rowB + q0 + mlane;
    const float* qp = qkv + qrow * (3 * Cc) + qoff;
#pragma unroll
    for (int f = 0; f < 2; ++f)
#pragma unroll
      for (int j = 0; j < 8; ++j) {
        qf[f][j]     = (__bf16)(qp[f * 32 + kbA + j] * scale);
        qf[f][j + 8] = (__bf16)(qp[f * 32 + kbA + 8 + j] * scale);
      }
  }

  v8f o[4];
#pragma unroll
  for (int t = 0; t < 4; ++t) o[t] = vz8();
  float mrow[8], lrow[8];
#pragma unroll
  for (int r = 0; r < 8; ++r) { mrow[r] = -1e30f; lrow[r] = 0.f; }

  for (int kt = 0; kt < q0 + 16; kt += 32) {
    // ---- stage V tile [32 keys][64 cols] -> LDS (coalesced float4) ----
#pragma unroll
    for (int i = 0; i < 16; ++i) {
      int idx4 = lane + i * 32;         // 512 float4 chunks
      int kr = idx4 >> 4;               // 16 chunks per key row
      int c4 = (idx4 & 15) * 4;
      const float4 v = *(const float4*)&qkv[(rowB + kt + kr) * (long)(3 * Cc) + voff + c4];
      vstg[wid][kr][c4 + 0] = (__bf16)v.x;
      vstg[wid][kr][c4 + 1] = (__bf16)v.y;
      vstg[wid][kr][c4 + 2] = (__bf16)v.z;
      vstg[wid][kr][c4 + 3] = (__bf16)v.w;
    }

    // ---- S(16q x 32k) = Q * K^T  (two 16-key subtiles) ----
    v8f s[2];
#pragma unroll
    for (int sub = 0; sub < 2; ++sub) {
      s[sub] = vz8();
      long krow = rowB + kt + sub * 16 + mlane;
      const float* kp = qkv + krow * (3 * Cc) + koff;
#pragma unroll
      for (int f = 0; f < 2; ++f) {
        v16bf kfrag;
#pragma unroll
        for (int j = 0; j < 16; ++j)
          kfrag[j] = (__bf16)kp[f * 32 + kbB + j];
        s[sub] = __builtin_amdgcn_wmma_f32_16x16x32_bf16(
            false, qf[f], false, kfrag, (short)0, s[sub], false, false);
      }
    }
    // ---- causal mask + online softmax (cross-lane in 16-lane halves) ----
#pragma unroll
    for (int r = 0; r < 8; ++r) {
      int q = q0 + r + rbase;
      if (kt + mlane      > q) s[0][r] = -1e30f;
      if (kt + 16 + mlane > q) s[1][r] = -1e30f;
      float t = fmaxf(s[0][r], s[1][r]);
      t = fmaxf(t, __shfl_xor(t, 1, 32));
      t = fmaxf(t, __shfl_xor(t, 2, 32));
      t = fmaxf(t, __shfl_xor(t, 4, 32));
      t = fmaxf(t, __shfl_xor(t, 8, 32));
      float mn = fmaxf(mrow[r], t);
      float sc = __expf(mrow[r] - mn);
      mrow[r] = mn;
      float p0 = __expf(s[0][r] - mn);
      float p1 = __expf(s[1][r] - mn);
      s[0][r] = p0; s[1][r] = p1;
      float rs = p0 + p1;
      rs += __shfl_xor(rs, 1, 32);
      rs += __shfl_xor(rs, 2, 32);
      rs += __shfl_xor(rs, 4, 32);
      rs += __shfl_xor(rs, 8, 32);
      lrow[r] = lrow[r] * sc + rs;
#pragma unroll
      for (int t4 = 0; t4 < 4; ++t4) o[t4][r] *= sc;
    }
    // ---- stage P (C-layout) then reload in A-layout; same-wave DS in-order ----
#pragma unroll
    for (int r = 0; r < 8; ++r) {
      pstg[wid][r + rbase][mlane]      = (__bf16)s[0][r];
      pstg[wid][r + rbase][16 + mlane] = (__bf16)s[1][r];
    }
    asm volatile("s_wait_dscnt 0" ::: "memory");
    v16bf pa;
#pragma unroll
    for (int j = 0; j < 8; ++j) {
      pa[j]     = pstg[wid][mlane][kbA + j];
      pa[j + 8] = pstg[wid][mlane][kbA + 8 + j];
    }
    // ---- O += P * V   (4 head-dim column tiles, V from LDS) ----
#pragma unroll
    for (int t = 0; t < 4; ++t) {
      v16bf vfrag;
#pragma unroll
      for (int j = 0; j < 16; ++j)
        vfrag[j] = vstg[wid][kbB + j][t * 16 + mlane];
      o[t] = __builtin_amdgcn_wmma_f32_16x16x32_bf16(
          false, pa, false, vfrag, (short)0, o[t], false, false);
    }
  }
  // ---- normalize + write ----
#pragma unroll
  for (int t = 0; t < 4; ++t)
#pragma unroll
    for (int r = 0; r < 8; ++r) {
      int q = q0 + r + rbase;
      y[(rowB + q) * (long)Cc + qoff + t * 16 + mlane] = o[t][r] / lrow[r];
    }
}

// ---------------------------------------------------------------------------
// RMSNorm: one block per row of C=1024
// ---------------------------------------------------------------------------
__global__ __launch_bounds__(256)
void rmsnorm_kernel(const float* __restrict__ x, const float* __restrict__ w,
                    float* __restrict__ out)
{
  __shared__ float red[8];
  const long row = blockIdx.x;
  const float* xr = x + row * Cc;
  float s = 0.f;
  for (int c = threadIdx.x; c < Cc; c += 256) { float v = xr[c]; s += v * v; }
  s += __shfl_xor(s, 1, 32);  s += __shfl_xor(s, 2, 32);
  s += __shfl_xor(s, 4, 32);  s += __shfl_xor(s, 8, 32);
  s += __shfl_xor(s, 16, 32);
  const int lane = threadIdx.x & 31, wid = threadIdx.x >> 5;
  if (lane == 0) red[wid] = s;
  __syncthreads();
  if (wid == 0) {
    float t = (lane < 8) ? red[lane] : 0.f;
    t += __shfl_xor(t, 1, 32); t += __shfl_xor(t, 2, 32); t += __shfl_xor(t, 4, 32);
    if (lane == 0) red[0] = t;
  }
  __syncthreads();
  const float inv = rsqrtf(red[0] / (float)Cc + 1e-6f);
  for (int c = threadIdx.x; c < Cc; c += 256)
    out[row * Cc + c] = xr[c] * inv * w[c];
}

// ---------------------------------------------------------------------------
// Router: softmax over E=8, top-2, normalized gates, loss stats
// ---------------------------------------------------------------------------
__global__ void router_kernel(const float* __restrict__ logits,
                              int* __restrict__ eflat, float* __restrict__ pflat,
                              float* __restrict__ cnt, float* __restrict__ imp)
{
  int i = blockIdx.x * blockDim.x + threadIdx.x;
  if (i >= Nrows) return;
  float l[Ee]; float mx = -1e30f;
#pragma unroll
  for (int e = 0; e < Ee; ++e) { l[e] = logits[(long)i * Ee + e]; mx = fmaxf(mx, l[e]); }
  float s = 0.f;
#pragma unroll
  for (int e = 0; e < Ee; ++e) { l[e] = __expf(l[e] - mx); s += l[e]; }
  const float inv = 1.f / s;
#pragma unroll
  for (int e = 0; e < Ee; ++e) l[e] *= inv;
  int i0 = 0; float v0 = l[0];
#pragma unroll
  for (int e = 1; e < Ee; ++e) if (l[e] > v0) { v0 = l[e]; i0 = e; }
  int i1 = -1; float v1 = -1.f;
#pragma unroll
  for (int e = 0; e < Ee; ++e) if (e != i0 && l[e] > v1) { v1 = l[e]; i1 = e; }
  atomicAdd(&cnt[i0], 1.f);
#pragma unroll
  for (int e = 0; e < Ee; ++e) atomicAdd(&imp[e], l[e]);
  const float den = v0 + v1 + 1e-9f;
  eflat[2 * i]     = i0;  pflat[2 * i]     = v0 / den;
  eflat[2 * i + 1] = i1;  pflat[2 * i + 1] = v1 / den;
}

// ---------------------------------------------------------------------------
// Stable rank-within-expert (== jax stable argsort semantics) -> slot id
// Single block, 256 threads, chunk = NKf/256
// ---------------------------------------------------------------------------
__global__ __launch_bounds__(256)
void order_kernel(const int* __restrict__ eflat, int* __restrict__ slot)
{
  __shared__ int cnts[256][Ee];
  const int t = threadIdx.x;
  const int chunk = NKf / 256;
  const int beg = t * chunk;
  int loc[Ee];
#pragma unroll
  for (int e = 0; e < Ee; ++e) loc[e] = 0;
  for (int i = 0; i < chunk; ++i) loc[eflat[beg + i]]++;
#pragma unroll
  for (int e = 0; e < Ee; ++e) cnts[t][e] = loc[e];
  __syncthreads();
  if (t < Ee) {
    int run = 0;
    for (int tt = 0; tt < 256; ++tt) { int v = cnts[tt][t]; cnts[tt][t] = run; run += v; }
  }
  __syncthreads();
  int base[Ee];
#pragma unroll
  for (int e = 0; e < Ee; ++e) base[e] = cnts[t][e];
  for (int i = 0; i < chunk; ++i) {
    int e = eflat[beg + i];
    int p = base[e]++;
    slot[beg + i] = (p < CAPp) ? (e * CAPp + p) : (Ee * CAPp);
  }
}

// ---------------------------------------------------------------------------
// Dispatch / combine / utility kernels
// ---------------------------------------------------------------------------
__global__ void scatter_kernel(const float* __restrict__ xn2, const int* __restrict__ slot,
                               const float* __restrict__ pflat,
                               float* __restrict__ xb, float* __restrict__ pb,
                               int* __restrict__ tb)
{
  const int i = blockIdx.x;            // flat (token,k) index
  const int s = slot[i];
  const int tok = i >> 1;
  const float* src = xn2 + (long)tok * Cc;
  float* dst = xb + (long)s * Cc;
  for (int c = threadIdx.x; c < Cc; c += blockDim.x) dst[c] = src[c];
  if (threadIdx.x == 0) { pb[s] = pflat[i]; tb[s] = tok; }
}

__global__ void combine_kernel(const float* __restrict__ outb, const float* __restrict__ pb,
                               const int* __restrict__ tb, float* __restrict__ out)
{
  const int r = blockIdx.x;            // slot row in [0, E*CAP)
  const float p = pb[r];
  if (p == 0.f) return;
  const int tok = tb[r];
  for (int c = threadIdx.x; c < Cc; c += blockDim.x)
    atomicAdd(&out[(long)tok * Cc + c], outb[(long)r * Cc + c] * p);
}

__global__ void fzero_kernel(float* p, long n) {
  long i = (long)blockIdx.x * blockDim.x + threadIdx.x;
  const long st = (long)gridDim.x * blockDim.x;
  for (; i < n; i += st) p[i] = 0.f;
}
__global__ void izero_kernel(int* p, long n) {
  long i = (long)blockIdx.x * blockDim.x + threadIdx.x;
  const long st = (long)gridDim.x * blockDim.x;
  for (; i < n; i += st) p[i] = 0;
}
__global__ void fcopy_kernel(float* __restrict__ d, const float* __restrict__ s, long n) {
  long i = (long)blockIdx.x * blockDim.x + threadIdx.x;
  const long st = (long)gridDim.x * blockDim.x;
  for (; i < n; i += st) d[i] = s[i];
}

__global__ void loss_kernel(const float* __restrict__ cnt, const float* __restrict__ imp,
                            float* __restrict__ out)
{
  float sc = 0.f, si = 0.f;
#pragma unroll
  for (int e = 0; e < Ee; ++e) { sc += cnt[e]; si += imp[e]; }
  float loss = 0.f;
#pragma unroll
  for (int e = 0; e < Ee; ++e)
    loss += (cnt[e] / (sc + 1e-9f)) * (imp[e] / (si + 1e-9f));
  out[0] = 0.01f * (float)Ee * loss;
}

// ---------------------------------------------------------------------------
// Host launcher
// ---------------------------------------------------------------------------
extern "C" void kernel_launch(void* const* d_in, const int* in_sizes, int n_in,
                              void* d_out, int out_size, void* d_ws, size_t ws_size,
                              hipStream_t stream)
{
  const float* x     = (const float*)d_in[0];
  const float* ln1   = (const float*)d_in[1];
  const float* wqkv  = (const float*)d_in[2];
  const float* wproj = (const float*)d_in[3];
  const float* ln2   = (const float*)d_in[4];
  const float* wgate = (const float*)d_in[5];
  const float* w1    = (const float*)d_in[6];
  const float* b1    = (const float*)d_in[7];
  const float* w2    = (const float*)d_in[8];
  const float* b2    = (const float*)d_in[9];
  float* out = (float*)d_out;

  // workspace carve-out (256B aligned regions)
  char* ws = (char*)d_ws;
  size_t off = 0;
  auto alloc = [&](size_t bytes) -> void* {
    void* p = ws + off;
    off = (off + bytes + 255) & ~(size_t)255;
    return p;
  };
  float* xn1    = (float*)alloc((size_t)Nrows * Cc * 4);
  float* qkvb   = (float*)alloc((size_t)Nrows * 3 * Cc * 4);
  float* yatt   = (float*)alloc((size_t)Nrows * Cc * 4);
  float* x1     = (float*)alloc((size_t)Nrows * Cc * 4);
  float* xn2    = (float*)alloc((size_t)Nrows * Cc * 4);
  float* logits = (float*)alloc((size_t)Nrows * Ee * 4);
  int*   eflat  = (int*)  alloc((size_t)NKf * 4);
  float* pflat  = (float*)alloc((size_t)NKf * 4);
  int*   slot   = (int*)  alloc((size_t)NKf * 4);
  float* cnt    = (float*)alloc(16 * 4);          // cnt[8] | imp[8]
  float* imp    = cnt + 8;
  float* xb     = (float*)alloc((size_t)(Ee * CAPp + 1) * Cc * 4);
  float* pb     = (float*)alloc((size_t)(Ee * CAPp + 1) * 4);
  int*   tb     = (int*)  alloc((size_t)(Ee * CAPp + 1) * 4);
  float* hbuf   = (float*)alloc((size_t)Ee * CAPp * Hh * 4);
  float* outb   = (float*)alloc((size_t)Ee * CAPp * Cc * 4);
  (void)ws_size; (void)in_sizes; (void)n_in; (void)out_size;

  // 1) xn1 = rmsnorm(x, ln1)
  rmsnorm_kernel<<<Nrows, 256, 0, stream>>>(x, ln1, xn1);
  // 2) qkv = xn1 @ w_qkv.T     (M=4096, N=3072, K=1024)
  gemm_wmma_kernel<false, false, true><<<dim3(3072 / BN, Nrows / BM, 1), 256, 0, stream>>>(
      xn1, wqkv, nullptr, nullptr, qkvb, Nrows, 3 * Cc, Cc, Cc, Cc, 3 * Cc, 0, 0, 0, 0);
  // 3) flash attention
  attn_kernel<<<dim3(Tt / 128, Bb * NHh), 256, 0, stream>>>(qkvb, yatt);
  // 4) x1 = x + yatt @ w_proj.T
  gemm_wmma_kernel<false, false, true><<<dim3(Cc / BN, Nrows / BM, 1), 256, 0, stream>>>(
      yatt, wproj, nullptr, x, x1, Nrows, Cc, Cc, Cc, Cc, Cc, 0, 0, 0, 0);
  // 5) xn2 = rmsnorm(x1, ln2)
  rmsnorm_kernel<<<Nrows, 256, 0, stream>>>(x1, ln2, xn2);
  // 6) gate logits = xn2 @ w_gate.T   (N=8, guarded tile)
  gemm_wmma_kernel<false, false, true><<<dim3(1, Nrows / BM, 1), 256, 0, stream>>>(
      xn2, wgate, nullptr, nullptr, logits, Nrows, Ee, Cc, Cc, Cc, Ee, 0, 0, 0, 0);
  // 7) zero routing state + dispatch buffers
  fzero_kernel<<<1, 32, 0, stream>>>(cnt, 16);
  fzero_kernel<<<2048, 256, 0, stream>>>(xb, (long)(Ee * CAPp + 1) * Cc);
  fzero_kernel<<<32, 256, 0, stream>>>(pb, Ee * CAPp + 1);
  izero_kernel<<<32, 256, 0, stream>>>(tb, Ee * CAPp + 1);
  // 8) router (softmax/top-2 + loss stats)
  router_kernel<<<Nrows / 256, 256, 0, stream>>>(logits, eflat, pflat, cnt, imp);
  // 9) stable capacity ranking
  order_kernel<<<1, 256, 0, stream>>>(eflat, slot);
  // 10) scatter tokens to expert buffers
  scatter_kernel<<<NKf, 128, 0, stream>>>(xn2, slot, pflat, xb, pb, tb);
  // 11) h = silu(xb @ w1[e] + b1[e])   batched over experts (B is [K,N])
  gemm_wmma_kernel<true, true, true><<<dim3((Hh + BN - 1) / BN, CAPp / BM, Ee), 256, 0, stream>>>(
      xb, w1, b1, nullptr, hbuf, CAPp, Hh, Cc, Cc, Hh, Hh,
      (long)CAPp * Cc, (long)Cc * Hh, Hh, (long)CAPp * Hh);
  // 12) outb = h @ w2[e] + b2[e]   (K=2730: unaligned A rows -> manual path)
  gemm_wmma_kernel<true, false, false><<<dim3(Cc / BN, CAPp / BM, Ee), 256, 0, stream>>>(
      hbuf, w2, b2, nullptr, outb, CAPp, Cc, Hh, Hh, Cc, Cc,
      (long)CAPp * Hh, (long)Hh * Cc, Cc, (long)CAPp * Cc);
  // 13) out = x1 (residual base)
  fcopy_kernel<<<2048, 256, 0, stream>>>(out, x1, (long)Nrows * Cc);
  // 14) out += gate-weighted expert outputs
  combine_kernel<<<Ee * CAPp, 256, 0, stream>>>(outb, pb, tb, out);
  // 15) load-balance loss scalar
  loss_kernel<<<1, 1, 0, stream>>>(cnt, imp, out + (long)Nrows * Cc);
}
